// MultiheadAttentionEinsum_24275155157745
// MI455X (gfx1250) — matrix operand order repaired and checked by reading, not compiled, not run on hardware
//
#include <hip/hip_runtime.h>
#include <hip/hip_bf16.h>

#define E_DIM 1024
#define HEADS 16
#define HDIM  64
#define BATCH 2
#define SEQ   2048

typedef __attribute__((ext_vector_type(16))) _Float16 v16h;
typedef __attribute__((ext_vector_type(8)))  _Float16 v8h;
typedef __attribute__((ext_vector_type(8)))  float    v8f;
typedef __attribute__((ext_vector_type(4)))  float    v4f;

static __device__ inline v16h cat16(v8h lo, v8h hi) {
  v16h r;
#pragma unroll
  for (int i = 0; i < 8; ++i) { r[i] = lo[i]; r[i + 8] = hi[i]; }
  return r;
}

static __device__ inline v8f wmma16(v16h a, v16h b, v8f c) {
  // D = A(16x32 f16) x B(32x16 f16) + C(16x16 f32)
  return __builtin_amdgcn_wmma_f32_16x16x32_f16(false, a, false, b, (short)0, c,
                                                false, false);
}

// ---------------------------------------------------------------------------
// Elementwise f32 -> f16 (one-time input conversion; all later reads hit L2)
// ---------------------------------------------------------------------------
__global__ __launch_bounds__(256)
void cvt_f16_kernel(const float* __restrict__ src, _Float16* __restrict__ dst) {
  const size_t i = ((size_t)blockIdx.x * 256 + threadIdx.x) * 8;
  v4f a = *(const v4f*)(src + i);
  v4f b = *(const v4f*)(src + i + 4);
  v8h o;
#pragma unroll
  for (int j = 0; j < 4; ++j) { o[j] = (_Float16)a[j]; o[j + 4] = (_Float16)b[j]; }
  *(v8h*)(dst + i) = o;
}

// ---------------------------------------------------------------------------
// One-time W[k][n] f32 -> Wt[n][k] f16 transpose via LDS (32x32 tiles)
// ---------------------------------------------------------------------------
__global__ __launch_bounds__(128)
void transpose_w_kernel(const float* __restrict__ W, _Float16* __restrict__ Wt) {
  __shared__ _Float16 T[32][33];
  const int tid = threadIdx.x;
  const int r0 = blockIdx.y * 32;   // k range
  const int c0 = blockIdx.x * 32;   // n range
  {
    const int row = tid >> 2;            // 0..31 (k)
    const int col = (tid & 3) * 8;       // 0,8,16,24 (n)
    const float* p = W + (size_t)(r0 + row) * E_DIM + c0 + col;
    v4f a = *(const v4f*)p;
    v4f b = *(const v4f*)(p + 4);
#pragma unroll
    for (int j = 0; j < 4; ++j) {
      T[row][col + j]     = (_Float16)a[j];
      T[row][col + 4 + j] = (_Float16)b[j];
    }
  }
  __syncthreads();
  {
    const int nrow = tid >> 2;           // 0..31 (n)
    const int kcol = (tid & 3) * 8;      // 0,8,16,24 (k)
    v8h o;
#pragma unroll
    for (int j = 0; j < 8; ++j) o[j] = T[kcol + j][nrow];
    *(v8h*)(Wt + (size_t)(c0 + nrow) * E_DIM + r0 + kcol) = o;
  }
}

// ---------------------------------------------------------------------------
// f16 GEMM: out_f16 = X_f16 @ W (given as Wt[n][k] f16) + bias(f32)
// Block = 4 waves; wave tile 32x64.  All 6 fragments loaded into distinct
// registers up front so WMMAs overlap subsequent loads (partial loadcnt waits).
// vlayout==0 -> out[(b*H+h)*T + t][d] (Q/K), ==1 -> out[(b*H+h)*D + d][t] (V^T)
// ---------------------------------------------------------------------------
__global__ __launch_bounds__(128)
void proj_kernel(const _Float16* __restrict__ X, const _Float16* __restrict__ Wt,
                 const float* __restrict__ bias, _Float16* __restrict__ out,
                 int vlayout) {
  const int tid  = threadIdx.x;
  const int lane = tid & 31;
  const int wave = tid >> 5;
  const int half = lane >> 4;
  const int l15  = lane & 15;
  const int m0   = blockIdx.x * 128;
  const int n0   = blockIdx.y * 64;

  v8f acc[2][4] = {};
  const int wm0 = m0 + wave * 32;
  const _Float16* Arow0 = X + (size_t)(wm0 + l15) * E_DIM;        // rows 0..15
  const _Float16* Arow1 = Arow0 + (size_t)16 * E_DIM;             // rows 16..31
  const _Float16* Brow  = Wt + (size_t)(n0 + l15) * E_DIM;

  for (int kk = 0; kk < E_DIM; kk += 32) {
    const int ka = kk + half * 8;
    const int kl = kk + half * 16;
    v16h a0 = cat16(*(const v8h*)(Arow0 + ka), *(const v8h*)(Arow0 + ka + 16));
    v16h a1 = cat16(*(const v8h*)(Arow1 + ka), *(const v8h*)(Arow1 + ka + 16));
    v16h bf[4];
#pragma unroll
    for (int c = 0; c < 4; ++c) {
      const _Float16* bp = Brow + (size_t)c * 16 * E_DIM + ka;
      bf[c] = cat16(*(const v8h*)bp, *(const v8h*)(bp + 16));
    }

    __builtin_prefetch(Arow0 + kk + 64, 0, 3);   // global_prefetch_b8
    __builtin_prefetch(Brow + kk + 64, 0, 3);

#pragma unroll
    for (int c = 0; c < 4; ++c) {
      acc[0][c] = wmma16(a0, bf[c], acc[0][c]);
      acc[1][c] = wmma16(a1, bf[c], acc[1][c]);
    }
  }

  // Epilogue: bias + head-major scatter
#pragma unroll
  for (int c = 0; c < 4; ++c) {
    const int n = n0 + c * 16 + l15;
    const float bv = bias[n];
    const int h = n >> 6, d = n & (HDIM - 1);
#pragma unroll
    for (int g = 0; g < 2; ++g)
#pragma unroll
      for (int r = 0; r < 8; ++r) {
        const int m = wm0 + g * 16 + r + half * 8;
        const int b = m >> 11, t = m & (SEQ - 1);
        const float val = acc[g][c][r] + bv;
        size_t addr;
        if (vlayout)
          addr = ((size_t)(b * HEADS + h) * HDIM + d) * SEQ + t;
        else
          addr = ((size_t)(b * HEADS + h) * SEQ + t) * HDIM + d;
        out[addr] = (_Float16)val;
      }
  }
}

// ---------------------------------------------------------------------------
// Flash attention: one wave per 16-query tile; keys in blocks of 32.
// K fragments loaded up front; V fragments issued before the softmax so L2
// latency hides under exp/shuffle VALU.  Same-wave LDS is in-order (DScnt),
// so the P-tile store->load bounce needs no explicit barrier.
// Q,K: [BH][T][64] f16.  VT: [BH][64][T] f16.  att: [B*T][E] f16.
// ---------------------------------------------------------------------------
__global__ __launch_bounds__(128)
void attn_kernel(const _Float16* __restrict__ Q, const _Float16* __restrict__ K,
                 const _Float16* __restrict__ VT, _Float16* __restrict__ att) {
  __shared__ _Float16 Pst[4][16][40] __attribute__((aligned(16)));
  const int tid  = threadIdx.x;
  const int lane = tid & 31;
  const int wave = tid >> 5;
  const int half = lane >> 4;
  const int l15  = lane & 15;

  const int bh = blockIdx.y;
  const int b  = bh / HEADS;
  const int h  = bh % HEADS;
  const int t0 = blockIdx.x * 64 + wave * 16;

  const _Float16* Qb = Q  + (size_t)bh * SEQ * HDIM;
  const _Float16* Kb = K  + (size_t)bh * SEQ * HDIM;
  const _Float16* Vb = VT + (size_t)bh * HDIM * SEQ;

  const _Float16* qrow = Qb + (size_t)(t0 + l15) * HDIM + half * 16;
  v16h qf0 = cat16(*(const v8h*)(qrow),      *(const v8h*)(qrow + 8));
  v16h qf1 = cat16(*(const v8h*)(qrow + 32), *(const v8h*)(qrow + 40));

  float mprev[8], lsum[8];
#pragma unroll
  for (int r = 0; r < 8; ++r) { mprev[r] = -1e30f; lsum[r] = 0.0f; }
  v8f O[4] = {};

  const float scale = 0.03125f;  // 1/sqrt(1024)

  for (int j0 = 0; j0 < SEQ; j0 += 32) {
    // ---- all global fragments for this key block, distinct registers ----
    v16h kb[4];
    {
      const _Float16* kp0 = Kb + (size_t)(j0 + l15) * HDIM + half * 16;
      const _Float16* kp1 = Kb + (size_t)(j0 + 16 + l15) * HDIM + half * 16;
      kb[0] = cat16(*(const v8h*)kp0,        *(const v8h*)(kp0 + 8));
      kb[1] = cat16(*(const v8h*)(kp0 + 32), *(const v8h*)(kp0 + 40));
      kb[2] = cat16(*(const v8h*)kp1,        *(const v8h*)(kp1 + 8));
      kb[3] = cat16(*(const v8h*)(kp1 + 32), *(const v8h*)(kp1 + 40));
    }
    v16h vfrag[4];
#pragma unroll
    for (int c = 0; c < 4; ++c) {
      const _Float16* vp = Vb + (size_t)(c * 16 + l15) * SEQ + j0 + half * 16;
      vfrag[c] = cat16(*(const v8h*)vp, *(const v8h*)(vp + 8));
    }

    v8f S0 = {}, S1 = {};
    S0 = wmma16(qf0, kb[0], S0);
    S0 = wmma16(qf1, kb[1], S0);
    S1 = wmma16(qf0, kb[2], S1);
    S1 = wmma16(qf1, kb[3], S1);

    // ---- online softmax update (rows live in 16-lane groups) ----
    float alpha[8];
#pragma unroll
    for (int r = 0; r < 8; ++r) {
      float s0 = S0[r] * scale, s1 = S1[r] * scale;
      float bm = fmaxf(s0, s1);
      bm = fmaxf(bm, __shfl_xor(bm, 1));
      bm = fmaxf(bm, __shfl_xor(bm, 2));
      bm = fmaxf(bm, __shfl_xor(bm, 4));
      bm = fmaxf(bm, __shfl_xor(bm, 8));
      const float mn = fmaxf(mprev[r], bm);
      const float a  = __expf(mprev[r] - mn);
      const float p0 = __expf(s0 - mn);
      const float p1 = __expf(s1 - mn);
      float ss = p0 + p1;
      ss += __shfl_xor(ss, 1);
      ss += __shfl_xor(ss, 2);
      ss += __shfl_xor(ss, 4);
      ss += __shfl_xor(ss, 8);
      lsum[r] = lsum[r] * a + ss;
      mprev[r] = mn;
      alpha[r] = a;
      // C-layout -> row-major P tile in this wave's private LDS region
      Pst[wave][r + half * 8][l15]      = (_Float16)p0;
      Pst[wave][r + half * 8][16 + l15] = (_Float16)p1;
    }
#pragma unroll
    for (int c = 0; c < 4; ++c)
#pragma unroll
      for (int r = 0; r < 8; ++r) O[c][r] *= alpha[r];

    // Reload P as a 16x32 f16 A-fragment (same-wave LDS is in-order)
    const _Float16* pp = &Pst[wave][l15][half * 16];
    v16h pfrag = cat16(*(const v8h*)pp, *(const v8h*)(pp + 8));

    // O += P @ V
#pragma unroll
    for (int c = 0; c < 4; ++c) O[c] = wmma16(pfrag, vfrag[c], O[c]);
  }

  // Normalize and write attended rows back to [B*T][E] f16
#pragma unroll
  for (int r = 0; r < 8; ++r) {
    const float inv = 1.0f / lsum[r];
    const int t = t0 + r + half * 8;
    const size_t rowbase = ((size_t)b * SEQ + t) * E_DIM + h * HDIM;
#pragma unroll
    for (int c = 0; c < 4; ++c)
      att[rowbase + c * 16 + l15] = (_Float16)(O[c][r] * inv);
  }
}

// ---------------------------------------------------------------------------
// Output projection: attended_f16 @ wo (as WoT[n][k] f16) + bo -> f32 d_out
// ---------------------------------------------------------------------------
__global__ __launch_bounds__(128)
void outproj_kernel(const _Float16* __restrict__ A, const _Float16* __restrict__ Wt,
                    const float* __restrict__ bias, float* __restrict__ out) {
  const int tid  = threadIdx.x;
  const int lane = tid & 31;
  const int wave = tid >> 5;
  const int half = lane >> 4;
  const int l15  = lane & 15;
  const int m0   = blockIdx.x * 128;
  const int n0   = blockIdx.y * 64;

  v8f acc[2][4] = {};
  const int wm0 = m0 + wave * 32;
  const _Float16* Arow0 = A + (size_t)(wm0 + l15) * E_DIM;
  const _Float16* Arow1 = Arow0 + (size_t)16 * E_DIM;
  const _Float16* Brow  = Wt + (size_t)(n0 + l15) * E_DIM;

  for (int kk = 0; kk < E_DIM; kk += 32) {
    const int ka = kk + half * 8;
    const int kl = kk + half * 16;
    v16h a0 = cat16(*(const v8h*)(Arow0 + ka), *(const v8h*)(Arow0 + ka + 16));
    v16h a1 = cat16(*(const v8h*)(Arow1 + ka), *(const v8h*)(Arow1 + ka + 16));
    v16h bf[4];
#pragma unroll
    for (int c = 0; c < 4; ++c) {
      const _Float16* bp = Brow + (size_t)c * 16 * E_DIM + ka;
      bf[c] = cat16(*(const v8h*)bp, *(const v8h*)(bp + 16));
    }

    __builtin_prefetch(Arow0 + kk + 64, 0, 3);
    __builtin_prefetch(Brow + kk + 64, 0, 3);

#pragma unroll
    for (int c = 0; c < 4; ++c) {
      acc[0][c] = wmma16(a0, bf[c], acc[0][c]);
      acc[1][c] = wmma16(a1, bf[c], acc[1][c]);
    }
  }

#pragma unroll
  for (int c = 0; c < 4; ++c) {
    const int n = n0 + c * 16 + l15;
    const float bv = bias[n];
#pragma unroll
    for (int g = 0; g < 2; ++g)
#pragma unroll
      for (int r = 0; r < 8; ++r) {
        const int m = wm0 + g * 16 + r + half * 8;
        out[(size_t)m * E_DIM + n] = acc[g][c][r] + bv;
      }
  }
}

extern "C" void kernel_launch(void* const* d_in, const int* in_sizes, int n_in,
                              void* d_out, int out_size, void* d_ws, size_t ws_size,
                              hipStream_t stream) {
  const float* query = (const float*)d_in[0];
  const float* key   = (const float*)d_in[1];
  const float* value = (const float*)d_in[2];
  const float* wq = (const float*)d_in[3];
  const float* bq = (const float*)d_in[4];
  const float* wk = (const float*)d_in[5];
  const float* bk = (const float*)d_in[6];
  const float* wv = (const float*)d_in[7];
  const float* bv = (const float*)d_in[8];
  const float* wo = (const float*)d_in[9];
  const float* bo = (const float*)d_in[10];
  float* out = (float*)d_out;

  const size_t QKV_ELEMS = (size_t)BATCH * HEADS * SEQ * HDIM;  // 4 Mi halves
  const size_t X_ELEMS   = (size_t)BATCH * SEQ * E_DIM;         // 4 Mi halves
  const size_t W_ELEMS   = (size_t)E_DIM * E_DIM;               // 1 Mi halves

  _Float16* p   = (_Float16*)d_ws;
  _Float16* qf  = p;  p += QKV_ELEMS;      // Q  [BH][T][D]
  _Float16* kf  = p;  p += QKV_ELEMS;      // K  [BH][T][D]
  _Float16* vf  = p;  p += QKV_ELEMS;      // V^T[BH][D][T]
  _Float16* att = p;  p += X_ELEMS;        // attended [B*T][E]
  _Float16* xq  = p;  p += X_ELEMS;        // f16 copies of inputs
  _Float16* xk  = p;  p += X_ELEMS;
  _Float16* xv  = p;  p += X_ELEMS;
  _Float16* wqT = p;  p += W_ELEMS;        // Wt[n][k] f16
  _Float16* wkT = p;  p += W_ELEMS;
  _Float16* wvT = p;  p += W_ELEMS;
  _Float16* woT = p;  p += W_ELEMS;

  // One-time conversions (everything then lives in the 192MB L2)
  const int cvtBlocks = (int)(X_ELEMS / (256 * 8));
  cvt_f16_kernel<<<cvtBlocks, 256, 0, stream>>>(query, xq);
  cvt_f16_kernel<<<cvtBlocks, 256, 0, stream>>>(key,   xk);
  cvt_f16_kernel<<<cvtBlocks, 256, 0, stream>>>(value, xv);

  dim3 gT(E_DIM / 32, E_DIM / 32);
  transpose_w_kernel<<<gT, 128, 0, stream>>>(wq, wqT);
  transpose_w_kernel<<<gT, 128, 0, stream>>>(wk, wkT);
  transpose_w_kernel<<<gT, 128, 0, stream>>>(wv, wvT);
  transpose_w_kernel<<<gT, 128, 0, stream>>>(wo, woT);

  dim3 gGemm((BATCH * SEQ) / 128, E_DIM / 64);
  proj_kernel<<<gGemm, 128, 0, stream>>>(xq, wqT, bq, qf, 0);
  proj_kernel<<<gGemm, 128, 0, stream>>>(xk, wkT, bk, kf, 0);
  proj_kernel<<<gGemm, 128, 0, stream>>>(xv, wvT, bv, vf, 1);

  attn_kernel<<<dim3(SEQ / 64, BATCH * HEADS), 128, 0, stream>>>(qf, kf, vf, att);

  outproj_kernel<<<gGemm, 128, 0, stream>>>(att, woT, bo, out);
}
